// RQVAE_23682449670812
// MI455X (gfx1250) — compile-verified, matrix operand order checked by reading, not run on hardware
//
#include <hip/hip_runtime.h>
#include <hip/hip_bf16.h>

// ---------------- problem constants (match reference) ----------------
#define NROWS 131072
#define DIN   384
#define HID   256
#define LDIM  32
#define KCB   512
#define QST   4

#define TILE_ROWS 128   // rows per block; 8 waves x 16-row strips
#define NBLOCKS   (NROWS / TILE_ROWS)   // 1024

typedef _Float16 half_t;
typedef __attribute__((ext_vector_type(16))) _Float16 v16h;
typedef __attribute__((ext_vector_type(8)))  _Float16 v8h;
typedef __attribute__((ext_vector_type(8)))  float    v8f;

// LDS layout (bytes). Dynamic LDS; total must be <= 320KB per workgroup.
#define LDS_XS   0                         // half [128][384]  = 98304
#define LDS_HA   (LDS_XS + 98304)          // half [128][256]  = 65536
#define LDS_HB   (LDS_HA + 65536)          // half [128][256]  = 65536
#define LDS_ZR   (LDS_HB + 65536)          // f32  [128][32]   = 16384  (residual)
#define LDS_ZQ   (LDS_ZR + 16384)          // f32  [128][32]   = 16384  (quant sum)
#define LDS_ZQB  (LDS_ZQ + 16384)          // half [128][32]   = 8192
#define LDS_RED  (LDS_ZQB + 8192)          // f32  [256]       = 1024
#define LDS_TOTAL (LDS_RED + 1024)         // 271360 bytes

#define SHUF16(lo, hi) __builtin_shufflevector(lo, hi, 0,1,2,3,4,5,6,7,8,9,10,11,12,13,14,15)

// ---------------------------------------------------------------------
// One wave computes a 16-row strip of  D = act(A*W^T + b)  via WMMA,
// processing TWO 16-col output tiles per iteration (shared A fragment,
// two accumulators, 4 B-fragment loads in flight per wait).
//   A : LDS half, [16][KD], row stride KD (this wave's strip base)
//   Wt: global half, [NO][KD] (pre-transposed weight, row = output col)
// MODE 0: relu -> half LDS out        (outH, ldoH)
// MODE 1: f32 LDS out + zero 2nd buf  (outF/ldoF = z residual, outF2 = qsum)
// MODE 2: f32 global out              (outF, ldoF)
// ---------------------------------------------------------------------
template<int KD, int NO, int MODE>
__device__ __forceinline__ void gemm16(const half_t* __restrict__ As,
                                       const half_t* __restrict__ Wt,
                                       const float*  __restrict__ bias,
                                       half_t* outH, int ldoH,
                                       float*  outF, int ldoF,
                                       float*  outF2,
                                       int lane)
{
    const int hl = lane & 15;   // half-lane: row of A / col of B within tile
    const int hi = lane >> 4;   // 0 or 1: selects K sub-phase per ISA layout
    const half_t* ap_base = As + hl * KD + hi * 8;

    for (int n0 = 0; n0 < NO; n0 += 32) {
        const half_t* bp0 = Wt + (size_t)(n0 + hl) * KD + hi * 8;
        const half_t* bp1 = bp0 + (size_t)16 * KD;
        v8f acc0 = {};
        v8f acc1 = {};
#pragma unroll
        for (int k0 = 0; k0 < KD; k0 += 32) {
            // A fragment (16-bit 16x32 layout): K = {base..+7, base+16..+23}
            v8h a0 = *(const v8h*)(ap_base + k0);
            v8h a1 = *(const v8h*)(ap_base + k0 + 16);
            v16h a = SHUF16(a0, a1);
            // two B fragments (adjacent 16-col tiles) -> 4 loads in flight
            v8h b00 = *(const v8h*)(bp0 + k0);
            v8h b01 = *(const v8h*)(bp0 + k0 + 16);
            v8h b10 = *(const v8h*)(bp1 + k0);
            v8h b11 = *(const v8h*)(bp1 + k0 + 16);
            v16h bA = SHUF16(b00, b01);
            v16h bB = SHUF16(b10, b11);
            acc0 = __builtin_amdgcn_wmma_f32_16x16x32_f16(
                       false, a, false, bA, (short)0, acc0, false, false);
            acc1 = __builtin_amdgcn_wmma_f32_16x16x32_f16(
                       false, a, false, bB, (short)0, acc1, false, false);
        }
        const float bn0 = bias[n0 + hl];
        const float bn1 = bias[n0 + 16 + hl];
#pragma unroll
        for (int e = 0; e < 8; ++e) {
            const int row = e + hi * 8;     // C/D layout: M = vgpr + 8*half
            float v0 = acc0[e] + bn0;
            float v1 = acc1[e] + bn1;
            if constexpr (MODE == 0) {
                v0 = v0 > 0.f ? v0 : 0.f;
                v1 = v1 > 0.f ? v1 : 0.f;
                outH[row * ldoH + n0 + hl]      = (half_t)v0;
                outH[row * ldoH + n0 + 16 + hl] = (half_t)v1;
            } else if constexpr (MODE == 1) {
                outF [row * ldoF + n0 + hl]      = v0;
                outF [row * ldoF + n0 + 16 + hl] = v1;
                outF2[row * ldoF + n0 + hl]      = 0.f;
                outF2[row * ldoF + n0 + 16 + hl] = 0.f;
            } else {
                outF[(size_t)row * ldoF + n0 + hl]      = v0;
                outF[(size_t)row * ldoF + n0 + 16 + hl] = v1;
            }
        }
    }
}

// -------- weight prep: Wt[n][k] = (half) W[k][n] -------------------------
__global__ void rqvae_transpose_h(const float* __restrict__ W,
                                  half_t* __restrict__ Wt, int Kd, int No)
{
    int i = blockIdx.x * blockDim.x + threadIdx.x;
    if (i >= Kd * No) return;
    int n = i / Kd;
    int k = i - n * Kd;
    Wt[i] = (half_t)W[(size_t)k * No + n];
}

// -------- fused encoder -> residual VQ -> decoder per 128-row tile -------
__global__ void __launch_bounds__(256, 1)
rqvae_fused(const float* __restrict__ x,
            const half_t* __restrict__ W1t, const float* __restrict__ b1,
            const half_t* __restrict__ W2t, const float* __restrict__ b2,
            const half_t* __restrict__ W3t, const float* __restrict__ b3,
            const half_t* __restrict__ W4t, const float* __restrict__ b4,
            const half_t* __restrict__ W5t, const float* __restrict__ b5,
            const half_t* __restrict__ W6t, const float* __restrict__ b6,
            const float* __restrict__ cb,
            float* __restrict__ out,
            float* __restrict__ idx_out,
            float* __restrict__ commit_partial)
{
    extern __shared__ char lds[];
    half_t* Xs   = (half_t*)(lds + LDS_XS);
    half_t* Ha   = (half_t*)(lds + LDS_HA);
    half_t* Hb   = (half_t*)(lds + LDS_HB);
    float*  Zres = (float*) (lds + LDS_ZR);
    float*  Zq   = (float*) (lds + LDS_ZQ);
    half_t* Zqb  = (half_t*)(lds + LDS_ZQB);
    float*  red  = (float*) (lds + LDS_RED);

    const int tid  = threadIdx.x;
    const int wave = tid >> 5;
    const int lane = tid & 31;
    const size_t row0 = (size_t)blockIdx.x * TILE_ROWS;
    const int strip = wave * 16;            // this wave's 16-row strip

    // ---- stage x tile into LDS as f16 (coalesced) ----
    for (int i = tid; i < TILE_ROWS * DIN; i += 256)
        Xs[i] = (half_t)x[row0 * DIN + i];
    __syncthreads();

    // ---- encoder ----
    gemm16<DIN, HID, 0>(Xs + strip * DIN, W1t, b1,
                        Ha + strip * HID, HID, nullptr, 0, nullptr, lane);
    __syncthreads();
    gemm16<HID, HID, 0>(Ha + strip * HID, W2t, b2,
                        Hb + strip * HID, HID, nullptr, 0, nullptr, lane);
    __syncthreads();
    gemm16<HID, LDIM, 1>(Hb + strip * HID, W3t, b3,
                         nullptr, 0, Zres + strip * LDIM, LDIM,
                         Zq + strip * LDIM, lane);
    __syncthreads();

    // ---- residual VQ (f32 VALU; each wave owns its 16 rows) ----
    float commit_local = 0.f;
    for (int q = 0; q < QST; ++q) {
        const float* cbq = cb + (size_t)q * KCB * LDIM;
        for (int r = 0; r < 16; ++r) {
            const int rl = strip + r;
            float rv[LDIM];
#pragma unroll
            for (int d = 0; d < LDIM; ++d) rv[d] = Zres[rl * LDIM + d];

            float best = 3.4e38f;
            int   bidx = KCB;
            for (int c = lane; c < KCB; c += 32) {        // 16 codewords/lane
                const float* cp = cbq + (size_t)c * LDIM;
                float dist = 0.f;
#pragma unroll
                for (int d = 0; d < LDIM; ++d) {
                    float t = rv[d] - cp[d];
                    dist += t * t;
                }
                if (dist < best) { best = dist; bidx = c; }
            }
            // wave argmin with lowest-index tie-break (matches jnp.argmin)
#pragma unroll
            for (int off = 16; off > 0; off >>= 1) {
                float ob = __shfl_xor(best, off, 32);
                int   oi = __shfl_xor(bidx, off, 32);
                if (ob < best || (ob == best && oi < bidx)) { best = ob; bidx = oi; }
            }
            if (lane == 0) {
                commit_local += best;   // best == ||qv - residual||^2
                idx_out[(row0 + (size_t)rl) * QST + q] = (float)bidx;
            }
            // 32 lanes update the 32 dims
            float qvd = cbq[(size_t)bidx * LDIM + lane];
            Zq  [rl * LDIM + lane] += qvd;
            Zres[rl * LDIM + lane] -= qvd;
        }
    }
    __syncthreads();

    // ---- z_q (== quant_sum under straight-through) to f16 ----
    for (int i = tid; i < TILE_ROWS * LDIM; i += 256)
        Zqb[i] = (half_t)Zq[i];
    __syncthreads();

    // ---- decoder ----
    gemm16<LDIM, HID, 0>(Zqb + strip * LDIM, W4t, b4,
                         Ha + strip * HID, HID, nullptr, 0, nullptr, lane);
    __syncthreads();
    gemm16<HID, HID, 0>(Ha + strip * HID, W5t, b5,
                        Hb + strip * HID, HID, nullptr, 0, nullptr, lane);
    __syncthreads();
    gemm16<HID, DIN, 2>(Hb + strip * HID, W6t, b6,
                        nullptr, 0, out + (row0 + strip) * DIN, DIN,
                        nullptr, lane);

    // ---- deterministic per-block commit partial ----
    red[tid] = commit_local;
    __syncthreads();
    if (tid == 0) {
        float s = 0.f;
        for (int i = 0; i < 256; ++i) s += red[i];
        commit_partial[blockIdx.x] = s;
    }
}

// -------- fixed-order final reduction of commit partials ----------------
__global__ void rqvae_commit_final(const float* __restrict__ partials,
                                   float* __restrict__ out_scalar)
{
    if (blockIdx.x == 0 && threadIdx.x == 0) {
        float s = 0.f;
        for (int i = 0; i < NBLOCKS; ++i) s += partials[i];
        *out_scalar = s * (1.0f / ((float)NROWS * (float)LDIM));
    }
}

// ------------------------------------------------------------------------
extern "C" void kernel_launch(void* const* d_in, const int* in_sizes, int n_in,
                              void* d_out, int out_size, void* d_ws, size_t ws_size,
                              hipStream_t stream)
{
    (void)in_sizes; (void)n_in; (void)out_size; (void)ws_size;

    const float* x      = (const float*)d_in[0];
    const float* enc_w1 = (const float*)d_in[1];
    const float* enc_b1 = (const float*)d_in[2];
    const float* enc_w2 = (const float*)d_in[3];
    const float* enc_b2 = (const float*)d_in[4];
    const float* enc_w3 = (const float*)d_in[5];
    const float* enc_b3 = (const float*)d_in[6];
    const float* dec_w1 = (const float*)d_in[7];
    const float* dec_b1 = (const float*)d_in[8];
    const float* dec_w2 = (const float*)d_in[9];
    const float* dec_b2 = (const float*)d_in[10];
    const float* dec_w3 = (const float*)d_in[11];
    const float* dec_b3 = (const float*)d_in[12];
    const float* cb     = (const float*)d_in[13];

    float* out      = (float*)d_out;                          // [N, DIN]
    float* idx_out  = out + (size_t)NROWS * DIN;              // [N, Q] as float
    float* commit_o = idx_out + (size_t)NROWS * QST;          // scalar

    // workspace carve (256B aligned regions)
    char* ws = (char*)d_ws;
    half_t* W1t = (half_t*)(ws + 0);        // [256][384] =  98304 el, 196608 B
    half_t* W2t = (half_t*)(ws + 196608);   // [256][256] =  65536 el, 131072 B
    half_t* W3t = (half_t*)(ws + 327680);   // [ 32][256] =   8192 el,  16384 B
    half_t* W4t = (half_t*)(ws + 344064);   // [256][ 32] =   8192 el,  16384 B
    half_t* W5t = (half_t*)(ws + 360448);   // [256][256] =  65536 el, 131072 B
    half_t* W6t = (half_t*)(ws + 491520);   // [384][256] =  98304 el, 196608 B
    float*  commit_partial = (float*)(ws + 688128);           // [NBLOCKS]

    // transpose+convert weights to f16 [Nout][K] (tiny; L2-resident afterward)
    auto tgrid = [](int n) { return (n + 255) / 256; };
    rqvae_transpose_h<<<tgrid(DIN*HID), 256, 0, stream>>>(enc_w1, W1t, DIN,  HID);
    rqvae_transpose_h<<<tgrid(HID*HID), 256, 0, stream>>>(enc_w2, W2t, HID,  HID);
    rqvae_transpose_h<<<tgrid(HID*LDIM),256, 0, stream>>>(enc_w3, W3t, HID,  LDIM);
    rqvae_transpose_h<<<tgrid(LDIM*HID),256, 0, stream>>>(dec_w1, W4t, LDIM, HID);
    rqvae_transpose_h<<<tgrid(HID*HID), 256, 0, stream>>>(dec_w2, W5t, HID,  HID);
    rqvae_transpose_h<<<tgrid(HID*DIN), 256, 0, stream>>>(dec_w3, W6t, HID,  DIN);

    // fused pipeline: 1024 blocks x 256 threads, ~265 KB dynamic LDS
    rqvae_fused<<<NBLOCKS, 256, LDS_TOTAL, stream>>>(
        x, W1t, enc_b1, W2t, enc_b2, W3t, enc_b3,
        W4t, dec_b1, W5t, dec_b2, W6t, dec_b3,
        cb, out, idx_out, commit_partial);

    rqvae_commit_final<<<1, 32, 0, stream>>>(commit_partial, commit_o);
}
